// PS_Axial_20169166422775
// MI455X (gfx1250) — matrix operand order repaired and checked by reading, not compiled, not run on hardware
//
#include <hip/hip_runtime.h>
#include <hip/hip_bf16.h>

typedef __attribute__((ext_vector_type(16))) __bf16 v16bf;
typedef __attribute__((ext_vector_type(8)))  __bf16 v8bf;
typedef __attribute__((ext_vector_type(8)))  float  v8f;

#define HW 3136          // 56*56
#define NBATCH 4
#define NBLK 196         // HW / 16
#define JC 64            // columns per chunk
#define NCHUNK (HW / JC) // 49

// ---------------------------------------------------------------------------
// Kernel 1: QKV projection (32x128 @ 128xHW) + folded BN.
// Emits bf16 feature rows:
//   Afeat[n][j][32] = [ q(8) | rq(8) | rk(8) | 0(8) ]   (i-side features)
//   Bfeat[n][j][32] = [ k(8) |  q(8) |  k(8) | 0(8) ]   (j-side features)
//   Vbf  [n][c][j]  = bf16( v[c,j] + rv[c,j] )          (channel-major for B-tiles)
// so that sc[i,j] = Afeat_row(i) . Bfeat_row(j) exactly.
// ---------------------------------------------------------------------------
__global__ __launch_bounds__(256) void PS_Axial_qkv_kernel(
    const float* __restrict__ x, const float* __restrict__ w_qkv,
    const float* __restrict__ b_qkv, const float* __restrict__ bn_gamma,
    const float* __restrict__ bn_beta, const float* __restrict__ bn_mean,
    const float* __restrict__ bn_var, const float* __restrict__ rq,
    const float* __restrict__ rk, const float* __restrict__ rv,
    __bf16* __restrict__ Afeat, __bf16* __restrict__ Bfeat,
    __bf16* __restrict__ Vbf)
{
    __shared__ float wl[32 * 128];
    __shared__ float sA[32];
    __shared__ float sB[32];

    const int tid = threadIdx.x;
    #pragma unroll
    for (int t = 0; t < 16; ++t)
        wl[tid + t * 256] = w_qkv[tid + t * 256];
    if (tid < 32) {
        float sc = bn_gamma[tid] * rsqrtf(bn_var[tid] + 1e-5f);
        sA[tid] = sc;
        sB[tid] = (b_qkv[tid] - bn_mean[tid]) * sc + bn_beta[tid];
    }
    __syncthreads();

    const int n = blockIdx.y;
    const int j = blockIdx.x * 256 + tid;
    if (j >= HW) return;

    const float* xb = x + (size_t)n * 128 * HW + j;
    float acc[32];
    #pragma unroll
    for (int o = 0; o < 32; ++o) acc[o] = 0.f;
    for (int c = 0; c < 128; ++c) {
        float xv = xb[(size_t)c * HW];
        #pragma unroll
        for (int o = 0; o < 32; ++o) acc[o] += xv * wl[o * 128 + c];
    }
    float val[32];
    #pragma unroll
    for (int o = 0; o < 32; ++o) val[o] = acc[o] * sA[o] + sB[o];

    v16bf alo, ahi, blo, bhi;
    #pragma unroll
    for (int c = 0; c < 8; ++c) {
        float q = val[c], k = val[8 + c];
        alo[c]     = (__bf16)q;
        alo[8 + c] = (__bf16)rq[c * HW + j];
        ahi[c]     = (__bf16)rk[c * HW + j];
        ahi[8 + c] = (__bf16)0.f;
        blo[c]     = (__bf16)k;
        blo[8 + c] = (__bf16)q;
        bhi[c]     = (__bf16)k;
        bhi[8 + c] = (__bf16)0.f;
    }
    v16bf* ap = (v16bf*)(Afeat + ((size_t)n * HW + j) * 32);
    ap[0] = alo; ap[1] = ahi;
    v16bf* bp = (v16bf*)(Bfeat + ((size_t)n * HW + j) * 32);
    bp[0] = blo; bp[1] = bhi;
    #pragma unroll
    for (int c = 0; c < 16; ++c)
        Vbf[((size_t)n * 16 + c) * HW + j] = (__bf16)(val[16 + c] + rv[c * HW + j]);
}

// ---------------------------------------------------------------------------
// Kernel 2: flash-style attention, software-pipelined.
// One wave32 per (batch, 16-row block); 64 columns per chunk (49 chunks).
// A-tile layout (16x32 bf16): lanes 0-15 rows M=0..15 hold K 0-15,
// lanes 16-31 hold K 16-31.  B-tile (32x16): lanes 0-15 cols N=0..15
// hold K 0-15, lanes 16-31 hold K 16-31.  C-tile f32: lane l, reg r ->
// (M = r + 8*(l>=16), N = l&15).
// Next chunk's B tiles are loaded before the softmax block so the global
// latency overlaps the ~16x64 exp/max VALU work.
// ---------------------------------------------------------------------------
__global__ __launch_bounds__(32) void PS_Axial_attn_kernel(
    const __bf16* __restrict__ Afeat, const __bf16* __restrict__ Bfeat,
    const __bf16* __restrict__ Vbf, float* __restrict__ out_un)
{
    __shared__ float  lbuf[16][JC + 1];  // logit tile, padded stride (65 dwords)
    __shared__ __bf16 pb[16][JC];        // softmaxed probabilities (bf16)
    __shared__ float  corrbuf[16];
    __shared__ float  sbuf[16];

    const int n    = blockIdx.x / NBLK;
    const int i0   = (blockIdx.x % NBLK) * 16;
    const int lane = threadIdx.x;
    const int half = lane >> 4;
    const int lid  = lane & 15;

    // Load the i-side A tile once (stays in registers for all chunks).
    const __bf16* abase = Afeat + ((size_t)n * HW + (i0 + lid)) * 32;
    v8bf a0 = *(const v8bf*)(abase + half * 16);
    v8bf a1 = *(const v8bf*)(abase + half * 16 + 8);
    v16bf amat;
    #pragma unroll
    for (int t = 0; t < 8; ++t) { amat[t] = a0[t]; amat[8 + t] = a1[t]; }

    const __bf16* brow = Bfeat + (size_t)n * HW * 32 + half * 16;
    const __bf16* vrow = Vbf + ((size_t)n * 16 + lid) * HW + half * 16;

    v8f acc = {};                 // PV accumulator (unnormalized)
    float m = -3.0e38f, s = 0.f;  // per-row running max/sum (lanes 0-15 own row=lid)

    // Preload B tiles for chunk 0.
    v16bf b[4];
    #pragma unroll
    for (int t = 0; t < 4; ++t)
        b[t] = *(const v16bf*)(brow + (size_t)(t * 16 + lid) * 32);

    for (int jc = 0; jc < NCHUNK; ++jc) {
        const int j0 = jc * JC;

        // Early independent loads: this chunk's V tiles (consumed after softmax).
        v16bf vb0 = *(const v16bf*)(vrow + j0);
        v16bf vb1 = *(const v16bf*)(vrow + j0 + 32);

        // --- QK: four 16x16 logit tiles, one WMMA each (K = 32 features) ---
        v8f z = {};
        v8f c[4];
        #pragma unroll
        for (int t = 0; t < 4; ++t)
            c[t] = __builtin_amdgcn_wmma_f32_16x16x32_bf16(false, amat, false, b[t],
                                                           (short)0, z, false, false);

        // Prefetch next chunk's B tiles; latency overlaps the softmax below.
        v16bf nb[4];
        if (jc + 1 < NCHUNK) {
            #pragma unroll
            for (int t = 0; t < 4; ++t)
                nb[t] = *(const v16bf*)(brow + (size_t)(j0 + JC + t * 16 + lid) * 32);
        }

        #pragma unroll
        for (int t = 0; t < 4; ++t)
            #pragma unroll
            for (int r = 0; r < 8; ++r)
                lbuf[r + 8 * half][t * 16 + lid] = c[t][r];
        __syncthreads();

        // --- online softmax: lanes 0-15 each own one row of the block ---
        if (half == 0) {
            float xv[JC];
            float cmax = -3.0e38f;
            #pragma unroll
            for (int k = 0; k < JC; ++k) { xv[k] = lbuf[lid][k]; cmax = fmaxf(cmax, xv[k]); }
            float mn   = fmaxf(m, cmax);
            float corr = __expf(m - mn);
            float csum = 0.f;
            #pragma unroll
            for (int k = 0; k < JC; ++k) {
                float p = __expf(xv[k] - mn);
                csum += p;
                pb[lid][k] = (__bf16)p;
            }
            s = s * corr + csum;
            m = mn;
            corrbuf[lid] = corr;
        }
        __syncthreads();

        // --- PV: P(16x64) @ V'(64x16) as two K=32 WMMAs, acc rescaled once ---
        v8bf p0 = *(const v8bf*)(&pb[lid][half * 16]);
        v8bf p1 = *(const v8bf*)(&pb[lid][half * 16 + 8]);
        v8bf p2 = *(const v8bf*)(&pb[lid][32 + half * 16]);
        v8bf p3 = *(const v8bf*)(&pb[lid][40 + half * 16]);
        v16bf pmat0, pmat1;
        #pragma unroll
        for (int t = 0; t < 8; ++t) {
            pmat0[t] = p0[t]; pmat0[8 + t] = p1[t];
            pmat1[t] = p2[t]; pmat1[8 + t] = p3[t];
        }

        v8f cin;
        #pragma unroll
        for (int r = 0; r < 8; ++r) cin[r] = acc[r] * corrbuf[r + 8 * half];
        cin = __builtin_amdgcn_wmma_f32_16x16x32_bf16(false, pmat0, false, vb0,
                                                      (short)0, cin, false, false);
        acc = __builtin_amdgcn_wmma_f32_16x16x32_bf16(false, pmat1, false, vb1,
                                                      (short)0, cin, false, false);
        __syncthreads();  // lbuf/pb reused next chunk

        #pragma unroll
        for (int t = 0; t < 4; ++t) b[t] = nb[t];
    }

    if (half == 0) sbuf[lid] = s;
    __syncthreads();

    // C-tile: lane -> channel c = lid, rows i0 + r + 8*half (8 consecutive floats)
    float* obase = out_un + ((size_t)n * 16 + lid) * HW + i0 + 8 * half;
    #pragma unroll
    for (int r = 0; r < 8; ++r) obase[r] = acc[r] / sbuf[r + 8 * half];
}

// ---------------------------------------------------------------------------
// Kernel 3: 2x2 average pool (4,16,56,56) -> (4,16,28,28)
// ---------------------------------------------------------------------------
__global__ __launch_bounds__(256) void PS_Axial_pool_kernel(
    const float* __restrict__ out_un, float* __restrict__ out)
{
    int idx = blockIdx.x * 256 + threadIdx.x;
    if (idx >= NBATCH * 16 * 28 * 28) return;
    int ow = idx % 28;
    int t  = idx / 28;
    int oh = t % 28;  t /= 28;
    int c  = t % 16;
    int n  = t / 16;
    const float* base = out_un + ((size_t)n * 16 + c) * HW + (2 * oh) * 56 + 2 * ow;
    out[idx] = 0.25f * (base[0] + base[1] + base[56] + base[57]);
}

// ---------------------------------------------------------------------------
extern "C" void kernel_launch(void* const* d_in, const int* in_sizes, int n_in,
                              void* d_out, int out_size, void* d_ws, size_t ws_size,
                              hipStream_t stream)
{
    (void)in_sizes; (void)n_in; (void)out_size; (void)ws_size;
    const float* x        = (const float*)d_in[0];
    const float* w_qkv    = (const float*)d_in[1];
    const float* b_qkv    = (const float*)d_in[2];
    const float* bn_gamma = (const float*)d_in[3];
    const float* bn_beta  = (const float*)d_in[4];
    const float* bn_mean  = (const float*)d_in[5];
    const float* bn_var   = (const float*)d_in[6];
    const float* rq       = (const float*)d_in[7];
    const float* rk       = (const float*)d_in[8];
    const float* rv       = (const float*)d_in[9];

    char* ws = (char*)d_ws;
    __bf16* Afeat  = (__bf16*)(ws);              // 4*3136*32*2 = 802816 B
    __bf16* Bfeat  = (__bf16*)(ws + 802816);     // 802816 B
    __bf16* Vbf    = (__bf16*)(ws + 1605632);    // 4*16*3136*2 = 401408 B
    float*  out_un = (float*)(ws + 2007040);     // 4*16*3136*4 = 802816 B

    PS_Axial_qkv_kernel<<<dim3((HW + 255) / 256, NBATCH), 256, 0, stream>>>(
        x, w_qkv, b_qkv, bn_gamma, bn_beta, bn_mean, bn_var, rq, rk, rv,
        Afeat, Bfeat, Vbf);

    PS_Axial_attn_kernel<<<NBATCH * NBLK, 32, 0, stream>>>(Afeat, Bfeat, Vbf, out_un);

    PS_Axial_pool_kernel<<<(NBATCH * 16 * 28 * 28 + 255) / 256, 256, 0, stream>>>(
        out_un, (float*)d_out);
}
